// BiLSTM_CRF_Dropout_25400436588899
// MI455X (gfx1250) — matrix operand order repaired and checked by reading, not compiled, not run on hardware
//
#include <hip/hip_runtime.h>
#include <hip/hip_bf16.h>
#include <math.h>

// Problem constants (match reference)
#define EDIM  128
#define HDIM  128
#define TTAG  9
#define BDIM  128
#define SDIM  256
#define G4H   (4 * HDIM)     // 512
#define ROWS  (SDIM * BDIM)  // 32768 time-major rows (r = s*B + b)
#define MTILES (ROWS / 16)   // 2048
#define NTILES (G4H / 16)    // 32
#define HPAD  (HDIM + 8)     // padded LDS row: 272 B -> conflict-free b128 loads

typedef __attribute__((ext_vector_type(16))) __bf16 v16bf;
typedef __attribute__((ext_vector_type(8)))  __bf16 v8bf;
typedef __attribute__((ext_vector_type(8)))  float  v8f;

__device__ __forceinline__ float sigmoidf_(float x) { return 1.0f / (1.0f + __expf(-x)); }
// tanh via hardware transcendentals: 1 - 2/(e^{2x}+1)
__device__ __forceinline__ float tanhf_(float x)    { return 1.0f - 2.0f / (__expf(2.0f * x) + 1.0f); }

// ---------------------------------------------------------------------------
// Kernel 1: fused embedding gather + input projection GEMM (both directions).
//   xp_frag[(mt*NTILES + nt)*32 + lane][v] = D fragment of
//     (embed[ids] @ w_ih^T + b_ih + b_hh) tile (mt, nt)
// Output stays in WMMA register layout so the LSTM kernel re-loads each gate
// accumulator with one contiguous v8f (2 x global_load_b128).
// ---------------------------------------------------------------------------
__global__ void __launch_bounds__(256)
proj_kernel(const int* __restrict__ ids, const float* __restrict__ embed,
            const float* __restrict__ w_ih_f, const float* __restrict__ b_ih_f,
            const float* __restrict__ b_hh_f,
            const float* __restrict__ w_ih_b, const float* __restrict__ b_ih_b,
            const float* __restrict__ b_hh_b,
            float* __restrict__ xp_f, float* __restrict__ xp_b)
{
    const int lane    = threadIdx.x & 31;
    const int halfSel = lane >> 4;      // 0 = lanes 0-15, 1 = lanes 16-31
    const int l16     = lane & 15;

    int wave = blockIdx.x * (blockDim.x >> 5) + (threadIdx.x >> 5);
    const int nt  = wave % NTILES;  wave /= NTILES;
    const int mt  = wave % MTILES;  wave /= MTILES;
    const int dir = wave;
    if (dir >= 2) return;

    const float* w_ih = dir ? w_ih_b : w_ih_f;
    float*       xp   = dir ? xp_b   : xp_f;

    // A-row for this lane: r = mt*16 + l16  ->  (s, b), gather embed row.
    const int row = mt * 16 + l16;
    const int s   = row / BDIM;
    const int b   = row % BDIM;
    const int id  = ids[b * SDIM + s];
    const float* arow = embed + (size_t)id * EDIM;

    // B column for this lane
    const int ncol = nt * 16 + l16;
    const float* brow = w_ih + (size_t)ncol * EDIM;   // w_ih[n][k], contiguous in k
    const float biasv = dir ? (b_ih_b[ncol] + b_hh_b[ncol])
                            : (b_ih_f[ncol] + b_hh_f[ncol]);

    v8f acc = {};
    #pragma unroll
    for (int kc = 0; kc < EDIM; kc += 32) {
        v16bf afr, bfr;
        // A k order used here (same as B): half0 -> K = kc+{0..15}; half1 -> K = kc+16+{0..15}
        const int ka0 = kc + halfSel * 16;
        #pragma unroll
        for (int i = 0; i < 16; ++i) afr[i]     = (__bf16)arow[ka0 + i];
        // B k order used here (same as A): half0 -> K = kc+{0..15}; half1 -> K = kc+16+{0..15}
        const int kb = kc + halfSel * 16;
        #pragma unroll
        for (int i = 0; i < 16; ++i) bfr[i] = (__bf16)brow[kb + i];

        acc = __builtin_amdgcn_wmma_f32_16x16x32_bf16(false, afr, false, bfr,
                                                      (short)0, acc, false, false);
    }

    // Store fragment contiguously (2 x global_store_b128)
    v8f outv;
    #pragma unroll
    for (int v = 0; v < 8; ++v) outv[v] = acc[v] + biasv;
    *(v8f*)(xp + ((size_t)(mt * NTILES + nt) * 32 + lane) * 8) = outv;
}

// ---------------------------------------------------------------------------
// Kernel 2: persistent bidirectional LSTM recurrence.
// grid.x = 16: blockIdx = dir*8 + batch-tile (16 batch rows each).
// block = 256 threads = 8 waves; wave w owns hidden tile w (16 hidden units)
// and computes the i,f,g,o gate tiles (nt = g*8 + w) for its batch tile.
// W_hh B-fragments resident in VGPRs; h double-buffered in padded LDS (bf16);
// c resident in accumulator registers. One barrier per step. WMMAs issued
// kc-outer / gate-inner so consecutive WMMAs are independent.
// ---------------------------------------------------------------------------
__global__ void __launch_bounds__(256)
lstm_kernel(const float* __restrict__ xp_f, const float* __restrict__ xp_b,
            const float* __restrict__ w_hh_f, const float* __restrict__ w_hh_b,
            float* __restrict__ h_f, float* __restrict__ h_b)
{
    const int lane    = threadIdx.x & 31;
    const int halfSel = lane >> 4;
    const int l16     = lane & 15;
    const int wv      = threadIdx.x >> 5;   // hidden tile 0..7
    const int dir     = blockIdx.x >> 3;
    const int mtile   = blockIdx.x & 7;     // batch rows mtile*16 .. +15

    const float* xp   = dir ? xp_b   : xp_f;
    const float* w_hh = dir ? w_hh_b : w_hh_f;
    float*       hout = dir ? h_b    : h_f;

    __shared__ __bf16 h_lds[2][16][HPAD];   // double-buffered, padded rows

    for (int i = threadIdx.x; i < 16 * HPAD; i += blockDim.x)
        (&h_lds[0][0][0])[i] = (__bf16)0.0f;

    // Resident W_hh fragments: 4 gates x 4 k-chunks (128 VGPRs of bf16)
    v16bf wfrag[4][4];
    #pragma unroll
    for (int g = 0; g < 4; ++g) {
        const int ncol = g * HDIM + wv * 16 + l16;
        const float* brow = w_hh + (size_t)ncol * HDIM;
        #pragma unroll
        for (int kc = 0; kc < 4; ++kc) {
            const int kb = kc * 32 + halfSel * 16;
            #pragma unroll
            for (int i = 0; i < 16; ++i) wfrag[g][kc][i] = (__bf16)brow[kb + i];
        }
    }

    v8f cfrag = {};     // cell state fragment (16 batch x 16 hidden), f32
    __syncthreads();

    for (int step = 0; step < SDIM; ++step) {
        const int s   = dir ? (SDIM - 1 - step) : step;
        const int cur = step & 1;
        const int nxt = cur ^ 1;
        const int rowTile = s * (BDIM / 16) + mtile;

        // Prefetch next step's xp fragments while this step computes
        if (step + 1 < SDIM) {
            const int sN = dir ? (s - 1) : (s + 1);
            const int rtN = sN * (BDIM / 16) + mtile;
            #pragma unroll
            for (int g = 0; g < 4; ++g) {
                const int nt = g * 8 + wv;
                __builtin_prefetch(xp + ((size_t)(rtN * NTILES + nt) * 32 + lane) * 8, 0, 1);
            }
        }

        // Load A fragments (h_prev) from LDS: 2 x ds_load_b128 per k-chunk,
        // conflict-free thanks to the 272 B row pitch.
        v16bf afrag[4];
        #pragma unroll
        for (int kc = 0; kc < 4; ++kc) {
            const int ka0 = kc * 32 + halfSel * 16;
            const int ka1 = kc * 32 + halfSel * 16 + 8;
            v8bf lo = *(const v8bf*)&h_lds[cur][l16][ka0];
            v8bf hi = *(const v8bf*)&h_lds[cur][l16][ka1];
            afrag[kc] = __builtin_shufflevector(lo, hi, 0, 1, 2, 3, 4, 5, 6, 7,
                                                8, 9, 10, 11, 12, 13, 14, 15);
        }

        // Gates = xp[s] + h_prev @ W_hh^T; accumulators pre-loaded with xp.
        v8f acc[4];
        #pragma unroll
        for (int g = 0; g < 4; ++g) {
            const int nt = g * 8 + wv;
            acc[g] = *(const v8f*)(xp + ((size_t)(rowTile * NTILES + nt) * 32 + lane) * 8);
        }
        // kc outer / g inner: 4 independent WMMA chains interleaved
        #pragma unroll
        for (int kc = 0; kc < 4; ++kc)
            #pragma unroll
            for (int g = 0; g < 4; ++g)
                acc[g] = __builtin_amdgcn_wmma_f32_16x16x32_bf16(
                    false, afrag[kc], false, wfrag[g][kc], (short)0, acc[g],
                    false, false);

        // Elementwise LSTM cell (torch gate order i,f,g,o), f32
        v8f hvec;
        #pragma unroll
        for (int v = 0; v < 8; ++v) {
            const float iv = sigmoidf_(acc[0][v]);
            const float fv = sigmoidf_(acc[1][v]);
            const float gv = tanhf_(acc[2][v]);
            const float ov = sigmoidf_(acc[3][v]);
            const float cv = fv * cfrag[v] + iv * gv;
            cfrag[v] = cv;
            const float hv = ov * tanhf_(cv);
            hvec[v] = hv;
            h_lds[nxt][halfSel * 8 + v][wv * 16 + l16] = (__bf16)hv;
        }
        // Global h in fragment layout (2 x global_store_b128)
        *(v8f*)(hout + ((size_t)((s * 8 + mtile) * 8 + wv) * 32 + lane) * 8) = hvec;

        __syncthreads();    // orders buf writes (step n) before buf reads (n+1)
    }
}

// ---------------------------------------------------------------------------
// Kernel 3: emission projection  em[b,s,t] = [hf|hb] . w_em[t] + b_em[t]
// h_f/h_b are in WMMA fragment layout: [(s*8+mtile)*8+wv][lane][v].
// ---------------------------------------------------------------------------
__global__ void __launch_bounds__(256)
emis_kernel(const float* __restrict__ h_f, const float* __restrict__ h_b,
            const float* __restrict__ w_em, const float* __restrict__ b_em,
            float* __restrict__ em)
{
    __shared__ float wsm[TTAG * 2 * HDIM];  // 9 KB
    __shared__ float bsm[TTAG];
    for (int i = threadIdx.x; i < TTAG * 2 * HDIM; i += blockDim.x) wsm[i] = w_em[i];
    if (threadIdx.x < TTAG) bsm[threadIdx.x] = b_em[threadIdx.x];
    __syncthreads();

    const int idx = blockIdx.x * blockDim.x + threadIdx.x;
    if (idx >= BDIM * SDIM) return;
    const int b = idx / SDIM;
    const int s = idx % SDIM;

    const int mtile   = b >> 4;
    const int lrow    = b & 15;
    const int halfSel = lrow >> 3;
    const int v       = lrow & 7;

    float out[TTAG];
    #pragma unroll
    for (int t = 0; t < TTAG; ++t) out[t] = bsm[t];

    for (int wv = 0; wv < 8; ++wv) {
        const size_t tileBase = ((size_t)((s * 8 + mtile) * 8 + wv) * 32 + halfSel * 16) * 8 + v;
        const float* pf = h_f + tileBase;
        const float* pb = h_b + tileBase;
        #pragma unroll
        for (int l = 0; l < 16; ++l) {
            const float vf = pf[(size_t)l * 8];
            const float vb = pb[(size_t)l * 8];
            const int h = wv * 16 + l;
            #pragma unroll
            for (int t = 0; t < TTAG; ++t)
                out[t] += vf * wsm[t * (2 * HDIM) + h] + vb * wsm[t * (2 * HDIM) + HDIM + h];
        }
    }
    #pragma unroll
    for (int t = 0; t < TTAG; ++t)
        em[((size_t)b * SDIM + s) * TTAG + t] = out[t];
}

// ---------------------------------------------------------------------------
// Kernel 4: CRF NLL. One block of B threads; thread b handles one sequence.
// ---------------------------------------------------------------------------
__global__ void __launch_bounds__(BDIM)
crf_kernel(const float* __restrict__ em, const int* __restrict__ tags,
           const float* __restrict__ start_trans, const float* __restrict__ end_trans,
           const float* __restrict__ trans, float* __restrict__ out)
{
    __shared__ float red[BDIM];
    __shared__ float tr[TTAG * TTAG];
    const int b = threadIdx.x;
    if (b < TTAG * TTAG) tr[b] = trans[b];
    __syncthreads();

    const float* e  = em   + (size_t)b * SDIM * TTAG;
    const int*   tg = tags + (size_t)b * SDIM;

    // Numerator: gold path score
    float score = start_trans[tg[0]] + e[tg[0]];
    for (int s = 1; s < SDIM; ++s)
        score += e[s * TTAG + tg[s]] + tr[tg[s - 1] * TTAG + tg[s]];
    score += end_trans[tg[SDIM - 1]];

    // Denominator: forward algorithm
    float alpha[TTAG];
    #pragma unroll
    for (int t = 0; t < TTAG; ++t) alpha[t] = start_trans[t] + e[t];
    for (int s = 1; s < SDIM; ++s) {
        float na[TTAG];
        #pragma unroll
        for (int j = 0; j < TTAG; ++j) {
            float m = -1e30f;
            #pragma unroll
            for (int i = 0; i < TTAG; ++i) m = fmaxf(m, alpha[i] + tr[i * TTAG + j]);
            float sum = 0.0f;
            #pragma unroll
            for (int i = 0; i < TTAG; ++i) sum += __expf(alpha[i] + tr[i * TTAG + j] - m);
            na[j] = m + __logf(sum) + e[s * TTAG + j];
        }
        #pragma unroll
        for (int j = 0; j < TTAG; ++j) alpha[j] = na[j];
    }
    float m = -1e30f;
    #pragma unroll
    for (int t = 0; t < TTAG; ++t) m = fmaxf(m, alpha[t] + end_trans[t]);
    float sum = 0.0f;
    #pragma unroll
    for (int t = 0; t < TTAG; ++t) sum += __expf(alpha[t] + end_trans[t] - m);
    const float logz = m + __logf(sum);

    red[b] = score - logz;
    __syncthreads();
    for (int off = BDIM / 2; off > 0; off >>= 1) {
        if (b < off) red[b] += red[b + off];
        __syncthreads();
    }
    if (b == 0) out[0] = -red[0] / (float)BDIM;
}

// ---------------------------------------------------------------------------
extern "C" void kernel_launch(void* const* d_in, const int* in_sizes, int n_in,
                              void* d_out, int out_size, void* d_ws, size_t ws_size,
                              hipStream_t stream) {
    const int*   input_ids   = (const int*)  d_in[0];
    const int*   tags        = (const int*)  d_in[1];
    const float* embed       = (const float*)d_in[2];
    const float* w_ih_f      = (const float*)d_in[3];
    const float* w_hh_f      = (const float*)d_in[4];
    const float* b_ih_f      = (const float*)d_in[5];
    const float* b_hh_f      = (const float*)d_in[6];
    const float* w_ih_b      = (const float*)d_in[7];
    const float* w_hh_b      = (const float*)d_in[8];
    const float* b_ih_b      = (const float*)d_in[9];
    const float* b_hh_b      = (const float*)d_in[10];
    const float* w_em        = (const float*)d_in[11];
    const float* b_em        = (const float*)d_in[12];
    const float* start_trans = (const float*)d_in[13];
    const float* end_trans   = (const float*)d_in[14];
    const float* trans       = (const float*)d_in[15];

    // Workspace layout (f32)
    float* ws  = (float*)d_ws;
    float* xpF = ws;                                   // ROWS * 4H (frag layout)
    float* xpB = xpF + (size_t)ROWS * G4H;
    float* hF  = xpB + (size_t)ROWS * G4H;             // S*B*H   (frag layout)
    float* hB  = hF  + (size_t)SDIM * BDIM * HDIM;
    float* em  = hB  + (size_t)SDIM * BDIM * HDIM;     // B*S*T

    // 1) input projections (both dirs): one wave per 16x16 tile
    {
        const int tiles = 2 * MTILES * NTILES;            // 131072
        const int wavesPerBlock = 8;
        const int blocks = tiles / wavesPerBlock;         // 16384
        proj_kernel<<<blocks, 256, 0, stream>>>(input_ids, embed,
                                                w_ih_f, b_ih_f, b_hh_f,
                                                w_ih_b, b_ih_b, b_hh_b,
                                                xpF, xpB);
    }
    // 2) bidirectional LSTM recurrence: 16 persistent blocks
    lstm_kernel<<<16, 256, 0, stream>>>(xpF, xpB, w_hh_f, w_hh_b, hF, hB);

    // 3) emissions
    emis_kernel<<<(BDIM * SDIM + 255) / 256, 256, 0, stream>>>(hF, hB, w_em, b_em, em);

    // 4) CRF NLL -> scalar
    crf_kernel<<<1, BDIM, 0, stream>>>(em, tags, start_trans, end_trans, trans,
                                       (float*)d_out);
}
